// CrossAttention_9921374454346
// MI455X (gfx1250) — matrix operand order repaired and checked by reading, not compiled, not run on hardware
//
#include <hip/hip_runtime.h>
#include <hip/hip_bf16.h>

typedef _Float16 f16;
typedef __attribute__((ext_vector_type(16))) _Float16 v16h;
typedef __attribute__((ext_vector_type(8)))  _Float16 v8h;
typedef __attribute__((ext_vector_type(8)))  float    v8f;
typedef __attribute__((ext_vector_type(4)))  int      v4i;

#define NB_   2
#define SQ_   2048
#define SKV_  2048
#define NH_   16
#define DH_   64
#define DM_   1024
#define DC_   768

#define WMMA(a,b,c) __builtin_amdgcn_wmma_f32_16x16x32_f16(false,(a),false,(b),(short)0,(c),false,false)

// ---- async copy to LDS (CDNA5) with safe fallback ----
#if __has_builtin(__builtin_amdgcn_global_load_async_to_lds_b128)
#define HAVE_ASYNC 1
typedef __attribute__((address_space(3))) v4i as3_v4i;
static __device__ inline void cp16_async(const void* g, void* l) {
  __builtin_amdgcn_global_load_async_to_lds_b128((v4i*)g, (as3_v4i*)l, 0, 0);
}
static __device__ inline void wait_async_all() {
  asm volatile("s_wait_asynccnt 0" ::: "memory");
}
#else
static __device__ inline void cp16_async(const void* g, void* l) {
  *(v8h*)l = *(const v8h*)g;
}
static __device__ inline void wait_async_all() {}
#endif

// A-fragment 16x32 (f16) from row-major [*, ld]: lane l -> row l%16,
// halves 0..7 = K kb..kb+7, halves 8..15 = K kb+16..kb+23, kb=(l<16?0:8).
static __device__ inline v16h load_frag_a(const f16* p, int ld) {
  const int lane = threadIdx.x & 31;
  const int row  = lane & 15;
  const int kb   = (lane & 16) ? 8 : 0;
  const f16* q = p + row * ld + kb;
  union { v16h v; v8h h[2]; } u;
  u.h[0] = *(const v8h*)(q);
  u.h[1] = *(const v8h*)(q + 16);
  return u.v;
}

// B-fragment 32x16 (f16), source stored [N, Kinner] row-major: lane l -> column
// N=l%16, halves 0..7 = K kb..kb+7, halves 8..15 = K kb+16..kb+23, kb=(l<16?0:8).
static __device__ inline v16h load_frag_b(const f16* p, int ld) {
  const int lane = threadIdx.x & 31;
  const int n    = lane & 15;
  const int kb   = (lane & 16) ? 8 : 0;
  const f16* q = p + n * ld + kb;
  union { v16h v; v8h h[2]; } u;
  u.h[0] = *(const v8h*)(q);
  u.h[1] = *(const v8h*)(q + 16);
  return u.v;
}

// ---------------- conversion kernels ----------------
__global__ __launch_bounds__(256) void cvt_f32_to_f16(const float* __restrict__ in,
                                                      f16* __restrict__ out, int n) {
  int i = blockIdx.x * blockDim.x + threadIdx.x;
  if (i < n) out[i] = (f16)in[i];
}

// w [K,N] f32 (row-major) -> wt [N,K] f16
__global__ __launch_bounds__(256) void transpose_cvt(const float* __restrict__ w,
                                                     f16* __restrict__ wt, int K, int N) {
  int n = blockIdx.x * 32 + (threadIdx.x & 31);
  int k = blockIdx.y * 8 + (threadIdx.x >> 5);
  if (n < N && k < K) wt[(size_t)n * K + k] = (f16)w[(size_t)k * N + n];
}

// ---------------- projection GEMM ----------------
// C[M, 1024] = A[M,K] (f16) * Bt[1024,K]^T, stored f16 in head layout:
//   layoutT==0 : out[b, h, s, d]   (Q and K)
//   layoutT==1 : out[b, h, d, s]   (V, transposed for AV B-fragments)
// block = 256 (8 waves, 4 along M x 2 along N); wave tile = 32(M) x 64(N).
// K loop manually unrolled x2 with ping-pong fragment sets (K/32 is even):
// each half prefetches the other set, so no register copies, loads overlap WMMAs.
__global__ __launch_bounds__(256) void gemm_proj(const f16* __restrict__ A,
                                                 const f16* __restrict__ Bt,
                                                 f16* __restrict__ out,
                                                 int M, int K, int S, int layoutT, float scale) {
  const int wave = threadIdx.x >> 5;
  const int lane = threadIdx.x & 31;
  const int wm = wave & 3, wn = wave >> 2;
  const int m0 = blockIdx.x * 128 + wm * 32;
  const int n0 = blockIdx.y * 128 + wn * 64;
  const f16* A0 = A + (size_t)m0 * K;
  const f16* A1 = A + (size_t)(m0 + 16) * K;
  const f16* B0 = Bt + (size_t)n0 * K;

  v8f acc[8] = {};
  v16h a0A = load_frag_a(A0, K);
  v16h a1A = load_frag_a(A1, K);
  v16h bA[4], bB[4];
  v16h a0B, a1B;
#pragma unroll
  for (int t = 0; t < 4; ++t) bA[t] = load_frag_b(B0 + (size_t)t * 16 * K, K);

  const int nsteps = K / 32;   // 32 or 24, always even
  for (int s = 0; s < nsteps; s += 2) {
    {   // prefetch set B at step s+1 (always in range)
      const int k0 = (s + 1) * 32;
      a0B = load_frag_a(A0 + k0, K);
      a1B = load_frag_a(A1 + k0, K);
#pragma unroll
      for (int t = 0; t < 4; ++t) bB[t] = load_frag_b(B0 + (size_t)t * 16 * K + k0, K);
    }
#pragma unroll
    for (int t = 0; t < 4; ++t) {
      acc[t]     = WMMA(a0A, bA[t], acc[t]);
      acc[4 + t] = WMMA(a1A, bA[t], acc[4 + t]);
    }
    if (s + 2 < nsteps) {   // prefetch set A at step s+2
      const int k0 = (s + 2) * 32;
      a0A = load_frag_a(A0 + k0, K);
      a1A = load_frag_a(A1 + k0, K);
#pragma unroll
      for (int t = 0; t < 4; ++t) bA[t] = load_frag_b(B0 + (size_t)t * 16 * K + k0, K);
    }
#pragma unroll
    for (int t = 0; t < 4; ++t) {
      acc[t]     = WMMA(a0B, bB[t], acc[t]);
      acc[4 + t] = WMMA(a1B, bB[t], acc[4 + t]);
    }
  }

  const int rowoff = (lane & 16) ? 8 : 0;
  const int nl = lane & 15;
#pragma unroll
  for (int i = 0; i < 2; ++i)
#pragma unroll
    for (int t = 0; t < 4; ++t)
#pragma unroll
      for (int r = 0; r < 8; ++r) {
        int m = m0 + i * 16 + r + rowoff;
        int n = n0 + t * 16 + nl;
        int bb = m / S, ss = m - bb * S;
        int h = n >> 6, d = n & 63;
        size_t addr = layoutT ? (((size_t)(bb * NH_ + h) * DH_ + d) * S + ss)
                              : (((size_t)(bb * NH_ + h) * S + ss) * DH_ + d);
        out[addr] = (f16)(acc[i * 4 + t][r] * scale);
      }
}

// ---------------- flash attention ----------------
// grid.x = NB_*NH_*(SQ_/64); block = 128 (4 waves); each wave owns 16 queries.
// K/V chunks (32 kv) are staged cooperatively into LDS with double-buffered
// async loads; all 4 waves consume the shared tiles via ds_load fragments.
__global__ __launch_bounds__(128) void flash_attn(const f16* __restrict__ Q,
                                                  const f16* __restrict__ K,
                                                  const f16* __restrict__ Vt,
                                                  f16* __restrict__ O) {
  __shared__ __align__(16) f16 Ks[2][32 * 64];   // [kv][d]
  __shared__ __align__(16) f16 Vs[2][64 * 32];   // [d][kv]
  __shared__ __align__(16) f16 Ps[4][16 * 32];   // per-wave prob scratch
  const int tid  = threadIdx.x;
  const int wave = tid >> 5;
  const int lane = tid & 31;
  const int bh = blockIdx.x >> 5;                 // SQ_/64 == 32 q-tiles per (b,h)
  const int qt = (blockIdx.x & 31) * 64 + wave * 16;

  const f16* Qb = Q + ((size_t)bh * SQ_ + qt) * DH_;
  const char* Kb = (const char*)(K + (size_t)bh * SKV_ * DH_);
  const char* Vb = (const char*)(Vt + (size_t)bh * DH_ * SKV_);

  const v16h qa0 = load_frag_a(Qb, DH_);
  const v16h qa1 = load_frag_a(Qb + 32, DH_);

  const int vrow = tid >> 2;            // 0..31 (V row = head dim)
  const int vcol = (tid & 3) * 16;      // 16B sub-chunk within 64B row segment

  auto issue = [&](int buf, int kv0) {
    char* kd = (char*)&Ks[buf][0];
    char* vd = (char*)&Vs[buf][0];
    // K chunk: rows kv0..kv0+31 are 4KB contiguous ([skv][64] layout)
    cp16_async(Kb + (size_t)kv0 * 128 + tid * 16,        kd + tid * 16);
    cp16_async(Kb + (size_t)kv0 * 128 + 2048 + tid * 16, kd + 2048 + tid * 16);
    // V chunk: 64 rows (d) x 64B segments at column kv0 of [d][skv]
    cp16_async(Vb + (size_t)vrow * (SKV_ * 2) + kv0 * 2 + vcol,        vd + vrow * 64 + vcol);
    cp16_async(Vb + (size_t)(vrow + 32) * (SKV_ * 2) + kv0 * 2 + vcol, vd + (vrow + 32) * 64 + vcol);
  };

  float mrow[8], lrow[8], alpha[8];
  v8f acc[4] = {};
#pragma unroll
  for (int r = 0; r < 8; ++r) { mrow[r] = -1e30f; lrow[r] = 0.f; }

  f16* sp = &Ps[wave][0];
  const int rowoff = (lane & 16) ? 8 : 0;
  const int col = lane & 15;

  issue(0, 0);
  wait_async_all();
  __syncthreads();

  const int NC = SKV_ / 32;
  for (int c = 0; c < NC; ++c) {
    const int buf = c & 1;
    if (c + 1 < NC) issue(buf ^ 1, (c + 1) * 32);   // overlap next fetch with compute

    const f16* Kc = &Ks[buf][0];
    const f16* Vc = &Vs[buf][0];

    v8f s0 = {}, s1 = {};
    s0 = WMMA(qa0, load_frag_b(Kc, 64), s0);
    s0 = WMMA(qa1, load_frag_b(Kc + 32, 64), s0);
    s1 = WMMA(qa0, load_frag_b(Kc + 16 * 64, 64), s1);
    s1 = WMMA(qa1, load_frag_b(Kc + 16 * 64 + 32, 64), s1);

#pragma unroll
    for (int r = 0; r < 8; ++r) {
      float v = fmaxf(s0[r], s1[r]);
#pragma unroll
      for (int off = 8; off >= 1; off >>= 1) v = fmaxf(v, __shfl_xor(v, off, 32));
      float mn = fmaxf(mrow[r], v);
      float al = __expf(mrow[r] - mn);
      float p0 = __expf(s0[r] - mn);
      float p1 = __expf(s1[r] - mn);
      s0[r] = p0; s1[r] = p1;
      float rs = p0 + p1;
#pragma unroll
      for (int off = 8; off >= 1; off >>= 1) rs += __shfl_xor(rs, off, 32);
      lrow[r] = lrow[r] * al + rs;
      mrow[r] = mn;
      alpha[r] = al;
    }
#pragma unroll
    for (int t = 0; t < 4; ++t)
#pragma unroll
      for (int r = 0; r < 8; ++r) acc[t][r] *= alpha[r];

    // D-layout probs -> LDS [16 q][32 kv], reload as an A-fragment
#pragma unroll
    for (int r = 0; r < 8; ++r) {
      int row = r + rowoff;
      sp[row * 32 + col] = (f16)s0[r];
      sp[row * 32 + col + 16] = (f16)s1[r];
    }
    asm volatile("s_wait_dscnt 0" ::: "memory");
    v16h pa = load_frag_a(sp, 32);

#pragma unroll
    for (int t = 0; t < 4; ++t)
      acc[t] = WMMA(pa, load_frag_b(Vc + t * 16 * 32, 32), acc[t]);

    wait_async_all();   // next chunk landed
    __syncthreads();    // all waves done with buf before it is reused
  }

  const int b = bh >> 4, h = bh & 15;
#pragma unroll
  for (int t = 0; t < 4; ++t) {
#pragma unroll
    for (int r = 0; r < 8; ++r) {
      int qg = qt + r + rowoff;
      int d = t * 16 + col;
      float val = acc[t][r] / lrow[r];
      O[((size_t)(b * SQ_ + qg)) * DM_ + h * DH_ + d] = (f16)val;
    }
  }
}

// ---------------- output projection (f32 out, +bias) ----------------
__global__ __launch_bounds__(256) void gemm_out(const f16* __restrict__ A,
                                                const f16* __restrict__ Bt,
                                                const float* __restrict__ bias,
                                                float* __restrict__ out) {
  const int wave = threadIdx.x >> 5;
  const int lane = threadIdx.x & 31;
  const int wm = wave & 3, wn = wave >> 2;
  const int m0 = blockIdx.x * 128 + wm * 32;
  const int n0 = blockIdx.y * 128 + wn * 64;
  const f16* A0 = A + (size_t)m0 * DM_;
  const f16* A1 = A + (size_t)(m0 + 16) * DM_;
  const f16* B0 = Bt + (size_t)n0 * DM_;

  v8f acc[8] = {};
  v16h a0A = load_frag_a(A0, DM_);
  v16h a1A = load_frag_a(A1, DM_);
  v16h bA[4], bB[4];
  v16h a0B, a1B;
#pragma unroll
  for (int t = 0; t < 4; ++t) bA[t] = load_frag_b(B0 + (size_t)t * 16 * DM_, DM_);

  const int nsteps = DM_ / 32;   // 32, even
  for (int s = 0; s < nsteps; s += 2) {
    {
      const int k0 = (s + 1) * 32;
      a0B = load_frag_a(A0 + k0, DM_);
      a1B = load_frag_a(A1 + k0, DM_);
#pragma unroll
      for (int t = 0; t < 4; ++t) bB[t] = load_frag_b(B0 + (size_t)t * 16 * DM_ + k0, DM_);
    }
#pragma unroll
    for (int t = 0; t < 4; ++t) {
      acc[t]     = WMMA(a0A, bA[t], acc[t]);
      acc[4 + t] = WMMA(a1A, bA[t], acc[4 + t]);
    }
    if (s + 2 < nsteps) {
      const int k0 = (s + 2) * 32;
      a0A = load_frag_a(A0 + k0, DM_);
      a1A = load_frag_a(A1 + k0, DM_);
#pragma unroll
      for (int t = 0; t < 4; ++t) bA[t] = load_frag_b(B0 + (size_t)t * 16 * DM_ + k0, DM_);
    }
#pragma unroll
    for (int t = 0; t < 4; ++t) {
      acc[t]     = WMMA(a0B, bB[t], acc[t]);
      acc[4 + t] = WMMA(a1B, bB[t], acc[4 + t]);
    }
  }

  const int rowoff = (lane & 16) ? 8 : 0;
  const int nl = lane & 15;
#pragma unroll
  for (int i = 0; i < 2; ++i)
#pragma unroll
    for (int t = 0; t < 4; ++t) {
      float bv = bias[n0 + t * 16 + nl];
#pragma unroll
      for (int r = 0; r < 8; ++r) {
        int m = m0 + i * 16 + r + rowoff;
        int n = n0 + t * 16 + nl;
        out[(size_t)m * DM_ + n] = acc[i * 4 + t][r] + bv;
      }
    }
}

extern "C" void kernel_launch(void* const* d_in, const int* in_sizes, int n_in,
                              void* d_out, int out_size, void* d_ws, size_t ws_size,
                              hipStream_t stream) {
  const float* x     = (const float*)d_in[0];
  const float* cond  = (const float*)d_in[1];
  const float* w_q   = (const float*)d_in[2];
  const float* w_k   = (const float*)d_in[3];
  const float* w_v   = (const float*)d_in[4];
  const float* w_out = (const float*)d_in[5];
  const float* b_out = (const float*)d_in[6];
  float* out = (float*)d_out;

  char* ws = (char*)d_ws;
  size_t off = 0;
  f16* x16 = (f16*)(ws + off); off += (size_t)NB_ * SQ_ * DM_ * 2;
  f16* c16 = (f16*)(ws + off); off += (size_t)NB_ * SKV_ * DC_ * 2;
  f16* wqT = (f16*)(ws + off); off += (size_t)DM_ * DM_ * 2;
  f16* wkT = (f16*)(ws + off); off += (size_t)DM_ * DC_ * 2;
  f16* wvT = (f16*)(ws + off); off += (size_t)DM_ * DC_ * 2;
  f16* woT = (f16*)(ws + off); off += (size_t)DM_ * DM_ * 2;
  f16* Qh  = (f16*)(ws + off); off += (size_t)NB_ * NH_ * SQ_ * DH_ * 2;
  f16* Kh  = (f16*)(ws + off); off += (size_t)NB_ * NH_ * SKV_ * DH_ * 2;
  f16* Vt  = (f16*)(ws + off); off += (size_t)NB_ * NH_ * DH_ * SKV_ * 2;
  f16* O16 = (f16*)(ws + off); off += (size_t)NB_ * SQ_ * DM_ * 2;

  {
    int nx = NB_ * SQ_ * DM_;
    cvt_f32_to_f16<<<nx / 256, 256, 0, stream>>>(x, x16, nx);
    int nc = NB_ * SKV_ * DC_;
    cvt_f32_to_f16<<<nc / 256, 256, 0, stream>>>(cond, c16, nc);
  }
  transpose_cvt<<<dim3(DM_ / 32, DM_ / 8), 256, 0, stream>>>(w_q, wqT, DM_, DM_);
  transpose_cvt<<<dim3(DM_ / 32, DC_ / 8), 256, 0, stream>>>(w_k, wkT, DC_, DM_);
  transpose_cvt<<<dim3(DM_ / 32, DC_ / 8), 256, 0, stream>>>(w_v, wvT, DC_, DM_);
  transpose_cvt<<<dim3(DM_ / 32, DM_ / 8), 256, 0, stream>>>(w_out, woT, DM_, DM_);

  gemm_proj<<<dim3((NB_ * SQ_) / 128, DM_ / 128), 256, 0, stream>>>(
      x16, wqT, Qh, NB_ * SQ_, DM_, SQ_, 0, 0.125f);
  gemm_proj<<<dim3((NB_ * SKV_) / 128, DM_ / 128), 256, 0, stream>>>(
      c16, wkT, Kh, NB_ * SKV_, DC_, SKV_, 0, 1.0f);
  gemm_proj<<<dim3((NB_ * SKV_) / 128, DM_ / 128), 256, 0, stream>>>(
      c16, wvT, Vt, NB_ * SKV_, DC_, SKV_, 1, 1.0f);

  flash_attn<<<NB_ * NH_ * (SQ_ / 64), 128, 0, stream>>>(Qh, Kh, Vt, O16);

  gemm_out<<<dim3((NB_ * SQ_) / 128, DM_ / 128), 256, 0, stream>>>(O16, woT, b_out, out);

  (void)in_sizes; (void)n_in; (void)out_size; (void)ws_size; (void)off;
}
